// DeepPolyReLU_40183714021525
// MI455X (gfx1250) — matrix operand order repaired and checked by reading, not compiled, not run on hardware
//
#include <hip/hip_runtime.h>
#include <hip/hip_bf16.h>

#define N 4096
#define BM 128
#define BN 128
#define BK 32
#define LSTRIDE 80                      // BK*2 bytes + 16B pad -> conflict-free ds_load_b128
#define ATILE_BYTES (BM * LSTRIDE)      // 10240 B per operand tile
#define STAGE_BYTES (2 * ATILE_BYTES)   // A + B = 20480 B per stage
#define NSTAGES (N / BK)                // 128

#if defined(__has_builtin)
#if __has_builtin(__builtin_amdgcn_global_load_async_to_lds_b128) && \
    __has_builtin(__builtin_amdgcn_s_wait_asynccnt)
#define USE_ASYNC_LDS 1
#endif
#endif

typedef __attribute__((ext_vector_type(8)))  __bf16 v8bf;
typedef __attribute__((ext_vector_type(16))) __bf16 v16bf;
typedef __attribute__((ext_vector_type(8)))  float  v8f;

// exact operand types for the async-to-LDS builtin (from clang's diagnostics):
// arg0: AS1 pointer to 16B int vector, arg1: AS3 pointer to 16B int vector,
// arg2/arg3: int immediates (offset, cpol)
typedef int async_v4i __attribute__((vector_size(4 * sizeof(int))));
typedef __attribute__((address_space(1))) async_v4i* async_gptr;
typedef __attribute__((address_space(3))) async_v4i* async_lptr;

__device__ __forceinline__ unsigned short f32_to_bf16_rne(float f) {
    unsigned int u = __float_as_uint(f);
    unsigned int r = u + 0x7FFFu + ((u >> 16) & 1u);
    return (unsigned short)(r >> 16);
}

// ---------------- zero fill ----------------
__global__ void zero_f4(float4* __restrict__ p, size_t n4) {
    size_t t = (size_t)blockIdx.x * blockDim.x + threadIdx.x;
    size_t stride = (size_t)gridDim.x * blockDim.x;
    float4 z = {0.f, 0.f, 0.f, 0.f};
    for (; t < n4; t += stride) p[t] = z;
}

__global__ void zero_f32(float* __restrict__ p, int n) {
    int t = blockIdx.x * blockDim.x + threadIdx.x;
    if (t < n) p[t] = 0.f;
}

// ---------------- W2 fp32 [N,N] -> bf16 row-major (A operand) ----------------
__global__ void convert_w2(const float2* __restrict__ W2, unsigned int* __restrict__ W2b) {
    size_t t = (size_t)blockIdx.x * blockDim.x + threadIdx.x;
    size_t total = (size_t)N * N / 2;
    size_t stride = (size_t)gridDim.x * blockDim.x;
    for (; t < total; t += stride) {
        float2 v = W2[t];
        unsigned int lo = f32_to_bf16_rne(v.x);
        unsigned int hi = f32_to_bf16_rne(v.y);
        W2b[t] = lo | (hi << 16);
    }
}

// ---------------- W1 fp32 [k][i] -> bf16 transposed [i][k] (B operand) ----------------
__global__ void convert_w1t(const float* __restrict__ W1, unsigned short* __restrict__ W1T) {
    __shared__ unsigned short tile[32][33];
    const int ibase = blockIdx.x * 32;
    const int kbase = blockIdx.y * 32;
    for (int r = threadIdx.y; r < 32; r += 8)
        tile[r][threadIdx.x] = f32_to_bf16_rne(W1[(size_t)(kbase + r) * N + ibase + threadIdx.x]);
    __syncthreads();
    for (int r = threadIdx.y; r < 32; r += 8)
        W1T[(size_t)(ibase + r) * N + kbase + threadIdx.x] = tile[threadIdx.x][r];
}

// ---------------- fused GEMM (S = W2 @ W1) + interval reduction ----------------
// Block tile 128x128, double-buffered LDS staging of A (rows of W2) and B
// (rows of W1^T = columns of W1) via async global->LDS copies (ASYNCcnt).
// 8 waves: wave_m in {0,1} picks 64 rows, wave_n in {0..3} picks 32 cols;
// each wave holds 4x2 WMMA accumulators across the full K loop, then folds S
// through the sign-dependent lb0/ub0 weighting into per-row lbounds/ubounds
// via shfl reduction + global fp32 atomics.
__global__ void __launch_bounds__(256) gemm_bounds(
    const unsigned short* __restrict__ W2b,   // bf16 [N][N] row-major
    const unsigned short* __restrict__ W1Tb,  // bf16 [N][N] = W1^T row-major
    const float* __restrict__ lb0, const float* __restrict__ ub0,
    float* __restrict__ lbounds, float* __restrict__ ubounds)
{
    __shared__ __align__(16) unsigned char smem[2 * STAGE_BYTES];   // 40 KB

    const int t    = threadIdx.x;
    const int lane = t & 31;
    const int wave = t >> 5;
    const int l    = lane & 15;   // A-row / B-col selector within a tile
    const int h    = lane >> 4;   // lane-half: K sub-range per ISA layout
    const int wave_m = wave >> 2; // 0..1 -> 64-row half
    const int wave_n = wave & 3;  // 0..3 -> 32-col strip

    const int rowBase = blockIdx.y * BM;
    const int colBase = blockIdx.x * BN;

    const __bf16* Ag = (const __bf16*)W2b;
    const __bf16* Bg = (const __bf16*)W1Tb;

    // stage copier: 512 A-chunks + 512 B-chunks of 16B per stage, 2 of each per thread
    auto issue_stage = [&](int buf, int kk) {
        const unsigned base = (unsigned)buf * STAGE_BYTES;
#pragma unroll
        for (int i = 0; i < 2; ++i) {
            const int c = t + i * 256;          // 0..511
            const int r = c >> 2;               // row (A) / col (B): 0..127
            const int o = c & 3;                // 16B chunk within the 64B K-slice
            const __bf16* gA = Ag + (((size_t)(rowBase + r)) << 12) + kk + o * 8;
            const __bf16* gB = Bg + (((size_t)(colBase + r)) << 12) + kk + o * 8;
            unsigned char* lA = &smem[base + (unsigned)r * LSTRIDE + (unsigned)o * 16];
            unsigned char* lB = &smem[base + ATILE_BYTES + (unsigned)r * LSTRIDE + (unsigned)o * 16];
#ifdef USE_ASYNC_LDS
            __builtin_amdgcn_global_load_async_to_lds_b128(
                (async_gptr)gA, (async_lptr)lA, 0, 0);
            __builtin_amdgcn_global_load_async_to_lds_b128(
                (async_gptr)gB, (async_lptr)lB, 0, 0);
#else
            *(uint4*)lA = *(const uint4*)gA;
            *(uint4*)lB = *(const uint4*)gB;
#endif
        }
    };

    v8f acc[4][2] = {};

    issue_stage(0, 0);
    int buf = 0;
    for (int s = 0; s < NSTAGES; ++s) {
#ifdef USE_ASYNC_LDS
        __builtin_amdgcn_s_wait_asynccnt(0);
#endif
        __syncthreads();                                    // stage `buf` ready for all waves
        if (s + 1 < NSTAGES) issue_stage(buf ^ 1, (s + 1) * BK);

        const unsigned aoff = (unsigned)buf * STAGE_BYTES;
        const unsigned boff = aoff + ATILE_BYTES;

        // B frags: element j <-> K = 16*h + j at column l  -> 32B contiguous per lane
        v16bf Bf[2];
#pragma unroll
        for (int nt = 0; nt < 2; ++nt) {
            const unsigned bo = boff + (unsigned)(wave_n * 32 + nt * 16 + l) * LSTRIDE + (unsigned)h * 32;
            v8bf b0 = *(const v8bf*)&smem[bo];
            v8bf b1 = *(const v8bf*)&smem[bo + 16];
            Bf[nt] = __builtin_shufflevector(b0, b1, 0,1,2,3,4,5,6,7,8,9,10,11,12,13,14,15);
        }
        // A frags: element j <-> K = 16*h + j at row l  -> 32B contiguous per lane
#pragma unroll
        for (int mt = 0; mt < 4; ++mt) {
            const unsigned ao = aoff + (unsigned)(wave_m * 64 + mt * 16 + l) * LSTRIDE + (unsigned)h * 32;
            v8bf a0 = *(const v8bf*)&smem[ao];
            v8bf a1 = *(const v8bf*)&smem[ao + 16];
            v16bf Af = __builtin_shufflevector(a0, a1, 0,1,2,3,4,5,6,7,8,9,10,11,12,13,14,15);
            acc[mt][0] = __builtin_amdgcn_wmma_f32_16x16x32_bf16(false, Af, false, Bf[0],
                                                                 (short)0, acc[mt][0], false, false);
            acc[mt][1] = __builtin_amdgcn_wmma_f32_16x16x32_bf16(false, Af, false, Bf[1],
                                                                 (short)0, acc[mt][1], false, false);
        }
        __syncthreads();    // all waves done reading `buf` before it is overwritten
        buf ^= 1;
    }

    // acc[mt][nt][r] = S[rowBase + wave_m*64 + mt*16 + 8*h + r][colBase + wave_n*32 + nt*16 + l]
    const float lbv0 = lb0[colBase + wave_n * 32 + l];
    const float ubv0 = ub0[colBase + wave_n * 32 + l];
    const float lbv1 = lb0[colBase + wave_n * 32 + 16 + l];
    const float ubv1 = ub0[colBase + wave_n * 32 + 16 + l];
#pragma unroll
    for (int mt = 0; mt < 4; ++mt) {
        float ls[8], us[8];
#pragma unroll
        for (int r = 0; r < 8; ++r) {
            const float s0 = acc[mt][0][r];
            const float s1 = acc[mt][1][r];
            ls[r] = (s0 > 0.f ? s0 * lbv0 : s0 * ubv0) + (s1 > 0.f ? s1 * lbv1 : s1 * ubv1);
            us[r] = (s0 > 0.f ? s0 * ubv0 : s0 * lbv0) + (s1 > 0.f ? s1 * ubv1 : s1 * lbv1);
        }
#pragma unroll
        for (int off = 1; off < 16; off <<= 1) {
#pragma unroll
            for (int r = 0; r < 8; ++r) {
                ls[r] += __shfl_xor(ls[r], off, 32);
                us[r] += __shfl_xor(us[r], off, 32);
            }
        }
        if (l == 0) {
#pragma unroll
            for (int r = 0; r < 8; ++r) {
                const int row = rowBase + wave_m * 64 + mt * 16 + h * 8 + r;
                unsafeAtomicAdd(&lbounds[row], ls[r]);
                unsafeAtomicAdd(&ubounds[row], us[r]);
            }
        }
    }
}

// ---------------- finalize: intercepts + DeepPoly ReLU + output writes ----------------
__global__ void __launch_bounds__(256) finalize(
    const float* __restrict__ raw_alpha, const float* __restrict__ W2,
    const float* __restrict__ b1, const float* __restrict__ b2,
    const float* __restrict__ lbounds, const float* __restrict__ ubounds,
    float* __restrict__ out)
{
    const int lane = threadIdx.x & 31;
    const int wave = threadIdx.x >> 5;
    const int j = blockIdx.x * 8 + wave;
    if (j >= N) return;

    const float* w2row = W2 + (size_t)j * N;
    float dot = 0.f;
    for (int k = lane; k < N; k += 32) dot += b1[k] * w2row[k];
#pragma unroll
    for (int off = 16; off > 0; off >>= 1) dot += __shfl_xor(dot, off, 32);

    if (lane == 0) {
        const float cint = dot + b2[j];  // clintercept == cuintercept
        const float lb = lbounds[j] + cint;
        const float ub = ubounds[j] + cint;
        const float alpha = 1.f / (1.f + __expf(-raw_alpha[j]));
        const float denom = ub - lb;
        const float relu_slope = (denom == 0.f) ? 0.f : (ub / denom);
        const float relu_intercept = (1.f - relu_slope) * ub;
        const bool below = (ub <= 0.f);
        const bool above = (lb >= 0.f);
        const bool crossing = !(below || above);
        const float base = above ? 1.f : 0.f;
        const float uslope = crossing ? relu_slope : base;
        const float uintercept = crossing ? relu_intercept : 0.f;
        const float lsv1 = crossing ? 0.f : base;
        const float lsv2 = crossing ? 1.f : base;
        const float lslope = alpha * lsv1 + (1.f - alpha) * lsv2;

        const size_t NN = (size_t)N * N;
        out[(size_t)j * N + j] = lslope;            // diag(lslope)
        // out[NN + j] stays 0                       // lintercept
        out[NN + N + (size_t)j * N + j] = uslope;   // diag(uslope)
        out[2 * NN + N + j] = uintercept;           // uintercept
    }
}

extern "C" void kernel_launch(void* const* d_in, const int* in_sizes, int n_in,
                              void* d_out, int out_size, void* d_ws, size_t ws_size,
                              hipStream_t stream) {
    (void)in_sizes; (void)n_in; (void)ws_size;
    const float* raw_alpha = (const float*)d_in[0];
    const float* lb0       = (const float*)d_in[1];
    const float* ub0       = (const float*)d_in[2];
    const float* W1        = (const float*)d_in[3];
    const float* b1        = (const float*)d_in[4];
    const float* W2        = (const float*)d_in[5];
    const float* b2        = (const float*)d_in[6];
    float* out = (float*)d_out;

    // workspace: W2 bf16 (32MB) | W1^T bf16 (32MB) | lbounds (16KB) | ubounds (16KB)
    unsigned short* W2b = (unsigned short*)d_ws;
    unsigned short* W1T = W2b + (size_t)N * N;
    float* lbounds = (float*)(W1T + (size_t)N * N);
    float* ubounds = lbounds + N;

    zero_f4<<<4096, 256, 0, stream>>>((float4*)out, (size_t)out_size / 4);
    zero_f32<<<(2 * N + 255) / 256, 256, 0, stream>>>(lbounds, 2 * N);

    convert_w2<<<8192, 256, 0, stream>>>((const float2*)W2, (unsigned int*)W2b);
    dim3 tgrid(N / 32, N / 32), tblock(32, 8);
    convert_w1t<<<tgrid, tblock, 0, stream>>>(W1, W1T);

    dim3 ggrid(N / BN, N / BM);
    gemm_bounds<<<ggrid, 256, 0, stream>>>(W2b, W1T, lb0, ub0, lbounds, ubounds);

    finalize<<<512, 256, 0, stream>>>(raw_alpha, W2, b1, b2, lbounds, ubounds, out);
}
